// tk_25975962206646
// MI455X (gfx1250) — matrix operand order repaired and checked by reading, not compiled, not run on hardware
//
#include <hip/hip_runtime.h>
#include <hip/hip_bf16.h>

#define BB    32
#define LQ    64
#define LD    512
#define EMB   300
#define EP    320   // EMB padded to multiple of 32
#define FH    100
#define FHP   128   // FF_HID padded
#define AD    32    // ATT_DIM
#define CDIM  96    // 3*ATT_DIM
#define NH    8
#define HD    4
#define NK    21

typedef __attribute__((ext_vector_type(16))) _Float16 v16h;
typedef __attribute__((ext_vector_type(8)))  _Float16 v8h;
typedef __attribute__((ext_vector_type(8)))  float    v8f;

__device__ __forceinline__ float wred(float v) {
#pragma unroll
  for (int off = 16; off > 0; off >>= 1) v += __shfl_xor(v, off, 32);
  return v;
}

// ---------------- weight prep: dst[n][k] = src[k][n] (f32 -> f16, zero pad) ----
__global__ void k_wprep(const float* __restrict__ src, _Float16* __restrict__ dst,
                        int K0, int N0, int Kp, int Np) {
  int idx = blockIdx.x * 256 + threadIdx.x;
  if (idx >= Kp * Np) return;
  int n = idx / Kp, k = idx % Kp;
  float v = (k < K0 && n < N0) ? src[k * N0 + n] : 0.0f;
  dst[idx] = (_Float16)v;
}

// ---------------- embed * mask -> f16 padded ---------------------------------
__global__ void k_embed(const float* __restrict__ emb, const float* __restrict__ mask,
                        _Float16* __restrict__ x) {
  int row = blockIdx.x;
  float m = mask[row];
  const float* er = emb + (long long)row * EMB;
  _Float16* xr = x + (long long)row * EP;
  for (int e = threadIdx.x; e < EP; e += blockDim.x)
    xr[e] = (e < EMB) ? (_Float16)(er[e] * m) : (_Float16)0.0f;
}

// ---------------- generic WMMA GEMM: C = A(MxK) * Bt(NxK)^T + bias -----------
// A row-major f16 (lda = padded K), Bt row-major f16 (ldb = padded K).
// Per-wave 16x32 output tile: one shared A fragment, two B fragments / two
// accumulator chains per k-step (2 v_wmma per 6 b128 loads, dual XDL chains).
template <bool RELU, bool OUTF32>
__global__ __launch_bounds__(32) void k_gemm(
    const _Float16* __restrict__ A, int lda, long long sA,
    const _Float16* __restrict__ Bt, int ldb, long long sB,
    const float* __restrict__ bias, int biasN,
    void* __restrict__ Cv, int ldc, long long sC, int ksteps) {
  const int lane = threadIdx.x;
  const int nt2 = blockIdx.x, mt = blockIdx.y, bz = blockIdx.z;
  const int l16 = lane & 15, sel = lane >> 4;
  // A fragment (16x32 f16): lane holds row m=lane%16; k = sel*16 + h (contiguous, matches B)
  const _Float16* ar = A + (long long)bz * sA + (long long)(mt * 16 + l16) * lda + sel * 16;
  // B fragment (32x16 f16): lane holds col n=lane%16, k = sel*16 + h -> contiguous row of Bt
  const _Float16* br0 = Bt + (long long)bz * sB + (long long)(nt2 * 32 + l16) * ldb + sel * 16;
  const _Float16* br1 = br0 + (long long)16 * ldb;
  v8f acc0 = {};
  v8f acc1 = {};
  for (int ks = 0; ks < ksteps; ++ks) {
    v8h alo = *(const v8h*)(ar);
    v8h ahi = *(const v8h*)(ar + 8);
    v16h bf0 = *(const v16h*)(br0);
    v16h bf1 = *(const v16h*)(br1);
    v16h af;
#pragma unroll
    for (int i = 0; i < 8; ++i) { af[i] = alo[i]; af[i + 8] = ahi[i]; }
    acc0 = __builtin_amdgcn_wmma_f32_16x16x32_f16(false, af, false, bf0, (short)0, acc0,
                                                  false, false);
    acc1 = __builtin_amdgcn_wmma_f32_16x16x32_f16(false, af, false, bf1, (short)0, acc1,
                                                  false, false);
    ar += 32; br0 += 32; br1 += 32;
  }
  // C/D layout: lanes 0-15: n=lane, rows r; lanes 16-31: n=lane-16, rows r+8
  const int ncol0 = nt2 * 32 + l16;
  const int ncol1 = ncol0 + 16;
  const int mbase = mt * 16 + sel * 8;
  float bv0 = 0.0f, bv1 = 0.0f;
  if (bias != nullptr) {
    if (ncol0 < biasN) bv0 = bias[ncol0];
    if (ncol1 < biasN) bv1 = bias[ncol1];
  }
  const long long cb = (long long)bz * sC;
#pragma unroll
  for (int r = 0; r < 8; ++r) {
    float v0 = acc0[r] + bv0;
    float v1 = acc1[r] + bv1;
    if (RELU) { v0 = fmaxf(v0, 0.0f); v1 = fmaxf(v1, 0.0f); }
    long long rowb = cb + (long long)(mbase + r) * ldc;
    if (OUTF32) {
      ((float*)Cv)[rowb + ncol0] = v0;
      ((float*)Cv)[rowb + ncol1] = v1;
    } else {
      ((_Float16*)Cv)[rowb + ncol0] = (_Float16)v0;
      ((_Float16*)Cv)[rowb + ncol1] = (_Float16)v1;
    }
  }
}

// ---------------- LayerNorm(out = LN(a + r)), ddof=1, wave per row -----------
__global__ void k_ln(const _Float16* __restrict__ a, const _Float16* __restrict__ r,
                     const float* __restrict__ g, const float* __restrict__ be,
                     _Float16* __restrict__ out, int rows) {
  int wid = (blockIdx.x * blockDim.x + threadIdx.x) >> 5;
  int lane = threadIdx.x & 31;
  if (wid >= rows) return;
  const _Float16* ar = a + (long long)wid * EP;
  const _Float16* rr = r + (long long)wid * EP;
  float s = 0.0f;
  for (int e = lane; e < EMB; e += 32) s += (float)ar[e] + (float)rr[e];
  float mean = wred(s) * (1.0f / EMB);
  float vs = 0.0f;
  for (int e = lane; e < EMB; e += 32) {
    float x = (float)ar[e] + (float)rr[e] - mean;
    vs += x * x;
  }
  float denom = sqrtf(wred(vs) * (1.0f / (EMB - 1))) + 1e-6f;
  _Float16* orow = out + (long long)wid * EP;
  for (int e = lane; e < EP; e += 32) {
    float o = 0.0f;
    if (e < EMB) {
      float x = (float)ar[e] + (float)rr[e];
      o = g[e] * (x - mean) / denom + be[e];
    }
    orow[e] = (_Float16)o;
  }
}

// ---------------- fused attention: head_dim=4, softmax(sims*m)*m renorm ------
__global__ void k_attn(const _Float16* __restrict__ c, const float* __restrict__ mask,
                       _Float16* __restrict__ o, int T, float invscale) {
  __shared__ _Float16 sk[512 * HD];
  __shared__ _Float16 sv[512 * HD];
  __shared__ float    sm[512];
  const int tid = threadIdx.x;
  const int b = blockIdx.y / NH, head = blockIdx.y % NH;
  const long long base = (long long)b * T;
  for (int i = tid; i < T; i += blockDim.x) {
    sm[i] = mask[base + i];
    const _Float16* row = c + (base + i) * CDIM;
#pragma unroll
    for (int j = 0; j < HD; ++j) {
      sk[i * HD + j] = row[AD + head * HD + j];
      sv[i * HD + j] = row[2 * AD + head * HD + j];
    }
  }
  __syncthreads();
  int t = blockIdx.x * blockDim.x + tid;
  if (t >= T) return;
  const _Float16* qr = c + (base + t) * CDIM + head * HD;
  float q0 = (float)qr[0] * invscale, q1 = (float)qr[1] * invscale;
  float q2 = (float)qr[2] * invscale, q3 = (float)qr[3] * invscale;
  float mx = -3.0e38f;
  for (int i = 0; i < T; ++i) {
    float s = (q0 * (float)sk[i*HD] + q1 * (float)sk[i*HD+1] +
               q2 * (float)sk[i*HD+2] + q3 * (float)sk[i*HD+3]) * sm[i];
    mx = fmaxf(mx, s);
  }
  float Z = 0.0f, Sm = 0.0f, a0 = 0, a1 = 0, a2 = 0, a3 = 0;
  for (int i = 0; i < T; ++i) {
    float s = (q0 * (float)sk[i*HD] + q1 * (float)sk[i*HD+1] +
               q2 * (float)sk[i*HD+2] + q3 * (float)sk[i*HD+3]) * sm[i];
    float e = __expf(s - mx);
    Z += e;
    float em = e * sm[i];
    Sm += em;
    a0 += em * (float)sv[i*HD];   a1 += em * (float)sv[i*HD+1];
    a2 += em * (float)sv[i*HD+2]; a3 += em * (float)sv[i*HD+3];
  }
  float invZ = 1.0f / Z;
  float d = Sm * invZ + 1e-13f;
  _Float16* orow = o + (base + t) * AD + head * HD;
  orow[0] = (_Float16)(a0 * invZ / d); orow[1] = (_Float16)(a1 * invZ / d);
  orow[2] = (_Float16)(a2 * invZ / d); orow[3] = (_Float16)(a3 * invZ / d);
}

// ---------------- mix with raw embedding, L2 normalize -> f16 ----------------
__global__ void k_mixnorm(const float* __restrict__ emb, const float* __restrict__ mask,
                          const _Float16* __restrict__ enc, const float* __restrict__ mixer,
                          _Float16* __restrict__ outn, int rows) {
  int wid = (blockIdx.x * blockDim.x + threadIdx.x) >> 5;
  int lane = threadIdx.x & 31;
  if (wid >= rows) return;
  float mix = mixer[0];
  float m = mask[wid];
  const float* er = emb + (long long)wid * EMB;
  const _Float16* cr = enc + (long long)wid * EP;
  float ss = 0.0f;
  for (int e = lane; e < EMB; e += 32) {
    float v = (mix * er[e] * m + (1.0f - mix) * (float)cr[e]) * m;
    ss += v * v;
  }
  float inv = 1.0f / (sqrtf(wred(ss)) + 1e-13f);
  _Float16* orow = outn + (long long)wid * EP;
  for (int e = lane; e < EP; e += 32) {
    float o = 0.0f;
    if (e < EMB) o = (mix * er[e] * m + (1.0f - mix) * (float)cr[e]) * m * inv;
    orow[e] = (_Float16)o;
  }
}

__global__ void k_zero(float* p, int n) {
  int i = blockIdx.x * 256 + threadIdx.x;
  if (i < n) p[i] = 0.0f;
}

// ---------------- RBF kernel pooling: wave per (b, q) ------------------------
__global__ __launch_bounds__(32) void k_pool(const float* __restrict__ cosm,
                                             const float* __restrict__ mq,
                                             const float* __restrict__ md,
                                             const float* __restrict__ nns,
                                             float* __restrict__ perk) {
  const int bq = blockIdx.x;
  const int b = bq / LQ;
  const int lane = threadIdx.x;
  float acc[NK];
#pragma unroll
  for (int k = 0; k < NK; ++k) acc[k] = 0.0f;
  const float* crow = cosm + (long long)bq * LD;
  const float* mdr = md + b * LD;
  for (int d = lane; d < LD; d += 32) {
    float cv = crow[d];
    float mm = mdr[d];
#pragma unroll
    for (int k = 0; k < NK; ++k) {
      float mu = (k == 0) ? 1.0f : (0.95f - 0.1f * (k - 1));
      float i2 = (k == 0) ? 5.0e5f : 50.0f;   // 1/(2*sigma^2)
      float e = cv - mu;
      acc[k] += __expf(-e * e * i2) * mm;
    }
  }
  float mqv = mq[bq];
  float nsc = nns[0];
#pragma unroll
  for (int k = 0; k < NK; ++k) {
    float v = acc[k];
#pragma unroll
    for (int off = 16; off > 0; off >>= 1) v += __shfl_xor(v, off, 32);
    if (lane == 0) {
      float pk = v * mqv;                       // includes mq factor of qd_mask
      float lp = __log2f(fmaxf(pk, 1e-10f)) * nsc;
      atomicAdd(&perk[b * NK + k], lp * mqv);
    }
  }
}

__global__ void k_final(const float* __restrict__ perk, const float* __restrict__ dw,
                        float* __restrict__ out) {
  int b = threadIdx.x;
  if (b < BB) {
    float s = 0.0f;
#pragma unroll
    for (int k = 0; k < NK; ++k) s += perk[b * NK + k] * dw[k];
    out[b] = s;
  }
}

extern "C" void kernel_launch(void* const* d_in, const int* in_sizes, int n_in,
                              void* d_out, int out_size, void* d_ws, size_t ws_size,
                              hipStream_t stream) {
  const float* q_embed   = (const float*)d_in[0];
  const float* d_embed   = (const float*)d_in[1];
  const float* mask_q    = (const float*)d_in[2];
  const float* mask_d    = (const float*)d_in[3];
  const float* mixer     = (const float*)d_in[4];
  const float* nn_scaler = (const float*)d_in[5];
  const float* ff_w1     = (const float*)d_in[6];
  const float* ff_b1     = (const float*)d_in[7];
  const float* ff_w2     = (const float*)d_in[8];
  const float* ff_b2     = (const float*)d_in[9];
  const float* ln1_g     = (const float*)d_in[10];
  const float* ln1_b     = (const float*)d_in[11];
  const float* att_w     = (const float*)d_in[12];
  const float* att_b     = (const float*)d_in[13];
  const float* out_w     = (const float*)d_in[14];
  const float* out_b     = (const float*)d_in[15];
  const float* ln2_g     = (const float*)d_in[16];
  const float* ln2_b     = (const float*)d_in[17];
  const float* dense_w   = (const float*)d_in[18];

  char* wsp = (char*)d_ws;
  auto alloc = [&](size_t bytes) -> void* {
    void* p = (void*)wsp;
    wsp += (bytes + 255) & ~(size_t)255;
    return p;
  };

  const int RQ = BB * LQ;   // 2048
  const int RD = BB * LD;   // 16384

  _Float16* w1t = (_Float16*)alloc((size_t)2 * FHP * EP * 2);
  _Float16* w2t = (_Float16*)alloc((size_t)2 * EP * FHP * 2);
  _Float16* awt = (_Float16*)alloc((size_t)2 * CDIM * EP * 2);
  _Float16* owt = (_Float16*)alloc((size_t)2 * EP * AD * 2);
  _Float16* X   = (_Float16*)alloc((size_t)RD * EP * 2);
  _Float16* HID = (_Float16*)alloc((size_t)RD * FHP * 2);
  _Float16* FF  = (_Float16*)alloc((size_t)RD * EP * 2);   // reused for PROJ
  _Float16* H   = (_Float16*)alloc((size_t)RD * EP * 2);
  _Float16* CC  = (_Float16*)alloc((size_t)RD * CDIM * 2);
  _Float16* O   = (_Float16*)alloc((size_t)RD * AD * 2);
  _Float16* DN  = (_Float16*)alloc((size_t)RD * EP * 2);
  _Float16* QN  = (_Float16*)alloc((size_t)RQ * EP * 2);
  float*    COS = (float*)alloc((size_t)BB * LQ * LD * 4);
  float*    PERK = (float*)alloc((size_t)BB * NK * 4);

  // ---- weight prep: transpose to (N x Kpad) f16 ----
  for (int l = 0; l < 2; ++l) {
    k_wprep<<<(FHP * EP + 255) / 256, 256, 0, stream>>>(
        ff_w1 + (size_t)l * EMB * FH, w1t + (size_t)l * FHP * EP, EMB, FH, EP, FHP);
    k_wprep<<<(EP * FHP + 255) / 256, 256, 0, stream>>>(
        ff_w2 + (size_t)l * FH * EMB, w2t + (size_t)l * EP * FHP, FH, EMB, FHP, EP);
    k_wprep<<<(CDIM * EP + 255) / 256, 256, 0, stream>>>(
        att_w + (size_t)l * EMB * CDIM, awt + (size_t)l * CDIM * EP, EMB, CDIM, EP, CDIM);
    k_wprep<<<(EP * AD + 255) / 256, 256, 0, stream>>>(
        out_w + (size_t)l * AD * EMB, owt + (size_t)l * EP * AD, AD, EMB, AD, EP);
  }

  const float invscale = 0.16439898730535726f;  // 1/sqrt(300//8) = 1/sqrt(37)

  auto run_enc = [&](const float* emb, const float* mask, int L, int rows) {
    k_embed<<<rows, 256, 0, stream>>>(emb, mask, X);
    for (int l = 0; l < 2; ++l) {
      // hid = relu(X @ w1 + b1)
      k_gemm<true, false><<<dim3(FHP / 32, rows / 16, 1), 32, 0, stream>>>(
          X, EP, 0, w1t + (size_t)l * FHP * EP, EP, 0, ff_b1 + l * FH, FH,
          HID, FHP, 0, EP / 32);
      // ff = hid @ w2 + b2
      k_gemm<false, false><<<dim3(EP / 32, rows / 16, 1), 32, 0, stream>>>(
          HID, FHP, 0, w2t + (size_t)l * EP * FHP, FHP, 0, ff_b2 + l * EMB, EMB,
          FF, EP, 0, FHP / 32);
      // h = LN(ff + x)
      k_ln<<<rows / 8, 256, 0, stream>>>(FF, X, ln1_g + l * EMB, ln1_b + l * EMB, H, rows);
      // c = h @ att_w + att_b
      k_gemm<false, false><<<dim3(CDIM / 32, rows / 16, 1), 32, 0, stream>>>(
          H, EP, 0, awt + (size_t)l * CDIM * EP, EP, 0, att_b + l * CDIM, CDIM,
          CC, CDIM, 0, EP / 32);
      // fused masked softmax attention
      k_attn<<<dim3((L + 127) / 128, BB * NH), 128, 0, stream>>>(CC, mask, O, L, invscale);
      // proj = o @ out_w + out_b   (reuse FF)
      k_gemm<false, false><<<dim3(EP / 32, rows / 16, 1), 32, 0, stream>>>(
          O, AD, 0, owt + (size_t)l * EP * AD, AD, 0, out_b + l * EMB, EMB,
          FF, EP, 0, AD / 32);
      // x = LN(proj + h)
      k_ln<<<rows / 8, 256, 0, stream>>>(FF, H, ln2_g + l * EMB, ln2_b + l * EMB, X, rows);
    }
  };

  // d encoder first so shared buffers can be reused for q
  run_enc(d_embed, mask_d, LD, RD);
  k_mixnorm<<<RD / 8, 256, 0, stream>>>(d_embed, mask_d, X, mixer, DN, RD);
  run_enc(q_embed, mask_q, LQ, RQ);
  k_mixnorm<<<RQ / 8, 256, 0, stream>>>(q_embed, mask_q, X, mixer, QN, RQ);

  // batched cosine GEMM: cos[b] = QN[b] (64x320) @ DN[b]^T (320x512), fp32 out
  k_gemm<false, true><<<dim3(LD / 32, LQ / 16, BB), 32, 0, stream>>>(
      QN, EP, (long long)LQ * EP, DN, EP, (long long)LD * EP, nullptr, 0,
      COS, LD, (long long)LQ * LD, EP / 32);

  k_zero<<<(BB * NK + 255) / 256, 256, 0, stream>>>(PERK, BB * NK);
  k_pool<<<BB * LQ, 32, 0, stream>>>(COS, mask_q, mask_d, nn_scaler, PERK);
  k_final<<<1, 32, 0, stream>>>(PERK, dense_w, (float*)d_out);
}
